// MS_DGCNN2_24764781429020
// MI455X (gfx1250) — compile-verified, matrix operand order checked by reading, not run on hardware
//
#include <hip/hip_runtime.h>
#include <math.h>

// ---------------------------------------------------------------------------
// MS-DGCNN2 forward for MI455X (gfx1250, wave32, WMMA f16 16x16x32 -> f32 acc)
// ---------------------------------------------------------------------------

typedef _Float16 h16;
typedef __attribute__((ext_vector_type(16))) _Float16 v16h;
typedef __attribute__((ext_vector_type(2)))  _Float16 h2v;
typedef __attribute__((ext_vector_type(8)))  float    v8f;

#define EPS_BN 1e-5f
#define SLOPE  0.2f
#define NPTS   2048
#define NBATCH 16

__device__ __forceinline__ float lrelu(float v) { return v > 0.f ? v : SLOPE * v; }

// Load a 16x32 f16 fragment (A or B operand) from a row-major [rows][ld] f16
// matrix.  Per CDNA5 ISA 7.12.2: lanes 0-15 hold K = {0..7, 16..23}, lanes
// 16-31 hold K = {8..15, 24..31}; element i of v16h sits in VGPR i/2.
__device__ __forceinline__ v16h load_frag(const h16* __restrict__ base, int row,
                                          int ld, int k0, int lane) {
  const h16* p = base + (size_t)row * ld + k0 + ((lane < 16) ? 0 : 8);
  v16h f;
#pragma unroll
  for (int i = 0; i < 4; ++i) {
    h2v lo = *(const h2v*)(p + 2 * i);
    h2v hi = *(const h2v*)(p + 16 + 2 * i);
    f[2 * i]     = lo[0]; f[2 * i + 1]     = lo[1];
    f[8 + 2 * i] = hi[0]; f[8 + 2 * i + 1] = hi[1];
  }
  return f;
}

// ---------------------------------------------------------------------------
// Generic WMMA GEMM: Y[M][Cout] = act(W[Cout][Cin] * X[M][Cin]^T), f16 out.
// One wave per 16x16 output tile. act=1 -> BN(g,b)+LeakyReLU epilogue.
// ---------------------------------------------------------------------------
__global__ void gemm_conv(const h16* __restrict__ W, const h16* __restrict__ X,
                          h16* __restrict__ Y, const float* __restrict__ g,
                          const float* __restrict__ bb, int Cout, int Cin,
                          int act, int nWaves) {
  int wave = (int)((blockIdx.x * blockDim.x + threadIdx.x) >> 5);
  if (wave >= nWaves) return;                 // wave-uniform, EXEC stays full
  int lane   = threadIdx.x & 31;
  int tilesO = Cout >> 4;
  int tn = wave / tilesO, to = wave % tilesO;
  int rowA = to * 16 + (lane & 15);
  int rowB = tn * 16 + (lane & 15);
  v8f acc = {};
  for (int k0 = 0; k0 < Cin; k0 += 32) {
    v16h a = load_frag(W, rowA, Cin, k0, lane);
    v16h b = load_frag(X, rowB, Cin, k0, lane);
    acc = __builtin_amdgcn_wmma_f32_16x16x32_f16(false, a, false, b,
                                                 (short)0, acc, false, false);
  }
  int n  = tn * 16 + (lane & 15);
  int ob = to * 16 + ((lane < 16) ? 0 : 8);
  const float inv = rsqrtf(1.0f + EPS_BN);
#pragma unroll
  for (int r = 0; r < 8; r += 2) {
    float y0 = acc[r], y1 = acc[r + 1];
    if (act) {
      y0 = lrelu(y0 * (g[ob + r]     * inv) + bb[ob + r]);
      y1 = lrelu(y1 * (g[ob + r + 1] * inv) + bb[ob + r + 1]);
    }
    h2v o2; o2[0] = (h16)y0; o2[1] = (h16)y1;
    *(h2v*)(Y + (size_t)n * Cout + ob + r) = o2;
  }
}

// conv5: Cin=512 B-operand sourced piecewise from x1|x2|x3|x4 (xcat never
// materialized).  K-tiles of 32 never cross sub-tensor boundaries.
__global__ void gemm_conv5(const h16* __restrict__ W, const h16* __restrict__ x1,
                           const h16* __restrict__ x2, const h16* __restrict__ x3,
                           const h16* __restrict__ x4, h16* __restrict__ Y,
                           const float* __restrict__ g, const float* __restrict__ bb,
                           int nWaves) {
  int wave = (int)((blockIdx.x * blockDim.x + threadIdx.x) >> 5);
  if (wave >= nWaves) return;
  int lane = threadIdx.x & 31;
  const int Cout = 1024, Cin = 512;
  int tn = wave >> 6, to = wave & 63;
  int rowA = to * 16 + (lane & 15);
  int rowB = tn * 16 + (lane & 15);
  v8f acc = {};
  for (int k0 = 0; k0 < Cin; k0 += 32) {
    const h16* src; int ld, kk;
    if (k0 < 64)       { src = x1; ld = 64;  kk = k0;       }
    else if (k0 < 128) { src = x2; ld = 64;  kk = k0 - 64;  }
    else if (k0 < 256) { src = x3; ld = 128; kk = k0 - 128; }
    else               { src = x4; ld = 256; kk = k0 - 256; }
    v16h a = load_frag(W, rowA, Cin, k0, lane);
    v16h b = load_frag(src, rowB, ld, kk, lane);
    acc = __builtin_amdgcn_wmma_f32_16x16x32_f16(false, a, false, b,
                                                 (short)0, acc, false, false);
  }
  int n  = tn * 16 + (lane & 15);
  int ob = to * 16 + ((lane < 16) ? 0 : 8);
  const float inv = rsqrtf(1.0f + EPS_BN);
#pragma unroll
  for (int r = 0; r < 8; r += 2) {
    float y0 = lrelu(acc[r]     * (g[ob + r]     * inv) + bb[ob + r]);
    float y1 = lrelu(acc[r + 1] * (g[ob + r + 1] * inv) + bb[ob + r + 1]);
    h2v o2; o2[0] = (h16)y0; o2[1] = (h16)y1;
    *(h2v*)(Y + (size_t)n * Cout + ob + r) = o2;
  }
}

// Gram per batch: G[b][n][m] = sum_c X[b*N+n][c] * X[b*N+m][c]  (f16 out).
__global__ void gram_gemm(const h16* __restrict__ X, h16* __restrict__ G,
                          int C, int nWaves) {
  int wave = (int)((blockIdx.x * blockDim.x + threadIdx.x) >> 5);
  if (wave >= nWaves) return;
  int lane = threadIdx.x & 31;
  int b   = wave >> 14;            // (2048/16)^2 = 16384 tiles per batch
  int rem = wave & 16383;
  int tn = rem >> 7, tm = rem & 127;
  const h16* Xb = X + (size_t)b * NPTS * C;
  int rowA = tn * 16 + (lane & 15);
  int rowB = tm * 16 + (lane & 15);
  v8f acc = {};
  for (int k0 = 0; k0 < C; k0 += 32) {
    v16h a  = load_frag(Xb, rowA, C, k0, lane);
    v16h bf = load_frag(Xb, rowB, C, k0, lane);
    acc = __builtin_amdgcn_wmma_f32_16x16x32_f16(false, a, false, bf,
                                                 (short)0, acc, false, false);
  }
  int m  = tm * 16 + (lane & 15);
  int nr = tn * 16 + ((lane < 16) ? 0 : 8);
  h16* Gb = G + (size_t)b * NPTS * NPTS;
#pragma unroll
  for (int r = 0; r < 8; ++r)
    Gb[(size_t)(nr + r) * NPTS + m] = (h16)acc[r];
}

// ---------------------------------------------------------------------------
// kNN (top-k by negative squared distance), thread-per-row insertion sort.
// Ties resolved toward smaller index (strict >, ascending scan).
// ---------------------------------------------------------------------------
__global__ void knn_xyz(const float* __restrict__ x, int* __restrict__ idx) {
  int t = blockIdx.x * blockDim.x + threadIdx.x;
  if (t >= NBATCH * NPTS) return;
  int b = t >> 11, n = t & (NPTS - 1);
  const float* xb = x + (size_t)b * 3 * NPTS;
  float cx = xb[n], cy = xb[NPTS + n], cz = xb[2 * NPTS + n];
  const int K = 30;
  float best[30]; int bi[30];
  for (int i = 0; i < K; ++i) { best[i] = -3.4e38f; bi[i] = 0; }
  for (int m = 0; m < NPTS; ++m) {
    float dx = xb[m] - cx, dy = xb[NPTS + m] - cy, dz = xb[2 * NPTS + m] - cz;
    float s = -(dx * dx + dy * dy + dz * dz);
    if (s > best[K - 1]) {
      int j = K - 1;
      while (j > 0 && s > best[j - 1]) { best[j] = best[j - 1]; bi[j] = bi[j - 1]; --j; }
      best[j] = s; bi[j] = m;
    }
  }
  int* o = idx + (size_t)t * 30;
  for (int i = 0; i < K; ++i) o[i] = bi[i];
}

// score = 2*G[n][m] - xx[m]  (per-row constant -xx[n] dropped; same ordering)
__global__ void knn_feat(const h16* __restrict__ G, const float* __restrict__ xx,
                         int* __restrict__ idx) {
  int t = blockIdx.x * blockDim.x + threadIdx.x;
  if (t >= NBATCH * NPTS) return;
  int b = t >> 11;
  const h16* row = G + (size_t)t * NPTS;
  const float* xb = xx + (size_t)b * NPTS;
  const int K = 20;
  float best[20]; int bi[20];
  for (int i = 0; i < K; ++i) { best[i] = -3.4e38f; bi[i] = 0; }
  for (int m = 0; m < NPTS; ++m) {
    float s = 2.f * (float)row[m] - xb[m];
    if (s > best[K - 1]) {
      int j = K - 1;
      while (j > 0 && s > best[j - 1]) { best[j] = best[j - 1]; bi[j] = bi[j - 1]; --j; }
      best[j] = s; bi[j] = m;
    }
  }
  int* o = idx + (size_t)t * 30;
  for (int i = 0; i < K; ++i) o[i] = bi[i];
}

__global__ void rownorm(const h16* __restrict__ X, float* __restrict__ xx,
                        int C, int M) {
  int p = blockIdx.x * blockDim.x + threadIdx.x;
  if (p >= M) return;
  const h16* r = X + (size_t)p * C;
  float s = 0.f;
  for (int c = 0; c < C; ++c) { float v = (float)r[c]; s += v * v; }
  xx[p] = s;
}

// ---------------------------------------------------------------------------
// Hierarchical features (6/9/7 ch) + conv1{l,b,c} + BN + LeakyReLU + max_k.
// Block per point, 64 threads = output channels.
// ---------------------------------------------------------------------------
__global__ void hier_conv1(const float* __restrict__ x, const int* __restrict__ idx,
                           const float* w1l, const float* g1l, const float* b1l,
                           const float* w1b, const float* g1b, const float* b1b,
                           const float* w1c, const float* g1c, const float* b1c,
                           h16* __restrict__ xm) {
  int p = blockIdx.x;
  int b = p >> 11, n = p & (NPTS - 1);
  int o = threadIdx.x;
  __shared__ float sc[3];
  __shared__ float nb[30][3];
  const float* xb = x + (size_t)b * 3 * NPTS;
  if (o < 3) sc[o] = xb[o * NPTS + n];
  if (o < 30) {
    int j = idx[(size_t)p * 30 + o];
    nb[o][0] = xb[j]; nb[o][1] = xb[NPTS + j]; nb[o][2] = xb[2 * NPTS + j];
  }
  __syncthreads();
  const float inv = rsqrtf(1.f + EPS_BN);
  float s0 = sc[0], s1 = sc[1], s2 = sc[2];
  { // scale 0: k=5, f = [rel, ctr]
    const float* w = w1l + o * 6;
    float sg = g1l[o] * inv, bi = b1l[o], mx = -3.4e38f;
    for (int kk = 0; kk < 5; ++kk) {
      float rx = nb[kk][0] - s0, ry = nb[kk][1] - s1, rz = nb[kk][2] - s2;
      float y = w[0]*rx + w[1]*ry + w[2]*rz + w[3]*s0 + w[4]*s1 + w[5]*s2;
      mx = fmaxf(mx, lrelu(y * sg + bi));
    }
    xm[(size_t)p * 192 + o] = (h16)mx;
  }
  { // scale 1: k=20, f = [rel, rel/(||rel||+1e-8), ctr]
    const float* w = w1b + o * 9;
    float sg = g1b[o] * inv, bi = b1b[o], mx = -3.4e38f;
    for (int kk = 0; kk < 20; ++kk) {
      float rx = nb[kk][0] - s0, ry = nb[kk][1] - s1, rz = nb[kk][2] - s2;
      float nrm = sqrtf(rx*rx + ry*ry + rz*rz) + 1e-8f;
      float in_ = 1.f / nrm;
      float y = w[0]*rx + w[1]*ry + w[2]*rz +
                w[3]*rx*in_ + w[4]*ry*in_ + w[5]*rz*in_ +
                w[6]*s0 + w[7]*s1 + w[8]*s2;
      mx = fmaxf(mx, lrelu(y * sg + bi));
    }
    xm[(size_t)p * 192 + 64 + o] = (h16)mx;
  }
  { // scale 2: k=30, f = [rel, ctr, ||rel||]
    const float* w = w1c + o * 7;
    float sg = g1c[o] * inv, bi = b1c[o], mx = -3.4e38f;
    for (int kk = 0; kk < 30; ++kk) {
      float rx = nb[kk][0] - s0, ry = nb[kk][1] - s1, rz = nb[kk][2] - s2;
      float d = sqrtf(rx*rx + ry*ry + rz*rz);
      float y = w[0]*rx + w[1]*ry + w[2]*rz + w[3]*s0 + w[4]*s1 + w[5]*s2 + w[6]*d;
      mx = fmaxf(mx, lrelu(y * sg + bi));
    }
    xm[(size_t)p * 192 + 128 + o] = (h16)mx;
  }
}

// Edge conv combine: y = P[idx] + Q, BN + LeakyReLU, max over k=20.
__global__ void edge_combine(const h16* __restrict__ P, const h16* __restrict__ Q,
                             const int* __restrict__ idx, const float* __restrict__ g,
                             const float* __restrict__ bb, h16* __restrict__ Xo,
                             int Cout) {
  int p = blockIdx.x;
  int o = threadIdx.x;
  int b = p >> 11;
  float q  = (float)Q[(size_t)p * Cout + o];
  float sg = g[o] * rsqrtf(1.f + EPS_BN), bi = bb[o];
  const int* ip = idx + (size_t)p * 30;
  float mx = -3.4e38f;
  for (int kk = 0; kk < 20; ++kk) {
    int j = ip[kk];
    float v = (float)P[((size_t)(b * NPTS + j)) * Cout + o] + q;
    mx = fmaxf(mx, lrelu(v * sg + bi));
  }
  Xo[(size_t)p * Cout + o] = (h16)mx;
}

// ---------------------------------------------------------------------------
// Weight prep and tail kernels
// ---------------------------------------------------------------------------
__global__ void cvt_f16(const float* __restrict__ s, h16* __restrict__ d, int n) {
  int t = blockIdx.x * blockDim.x + threadIdx.x;
  if (t < n) d[t] = (h16)s[t];
}

// W[Cout][2Ci] -> Wd = f16(W[:, :Ci]),  Wdiff = f16(W[:, Ci:] - W[:, :Ci])
__global__ void split_w(const float* __restrict__ W, h16* __restrict__ Wd,
                        h16* __restrict__ Wf, int Cout, int Ci) {
  int t = blockIdx.x * blockDim.x + threadIdx.x;
  if (t >= Cout * Ci) return;
  int o = t / Ci, c = t % Ci;
  float a = W[(size_t)o * 2 * Ci + c];
  float d = W[(size_t)o * 2 * Ci + Ci + c];
  Wd[t] = (h16)a;
  Wf[t] = (h16)(d - a);
}

__global__ void pool_kernel(const h16* __restrict__ x5, float* __restrict__ h) {
  int t = blockIdx.x * blockDim.x + threadIdx.x;
  if (t >= NBATCH * 1024) return;
  int b = t >> 10, e = t & 1023;
  float mx = -3.4e38f, sm = 0.f;
  for (int n = 0; n < NPTS; ++n) {
    float v = (float)x5[((size_t)(b * NPTS + n)) * 1024 + e];
    mx = fmaxf(mx, v); sm += v;
  }
  h[(size_t)b * 2048 + e]        = mx;
  h[(size_t)b * 2048 + 1024 + e] = sm * (1.f / (float)NPTS);
}

__global__ void fc_kernel(const float* __restrict__ in, const float* __restrict__ W,
                          const float* __restrict__ bias, const float* __restrict__ g,
                          const float* __restrict__ bb, float* __restrict__ out,
                          int Bn, int In, int Out, int act) {
  int t = blockIdx.x * blockDim.x + threadIdx.x;
  if (t >= Bn * Out) return;
  int b = t / Out, o = t % Out;
  const float* ir = in + (size_t)b * In;
  const float* wr = W + (size_t)o * In;
  float acc = 0.f;
  for (int c = 0; c < In; ++c) acc += ir[c] * wr[c];
  if (bias) acc += bias[o];
  if (act)  acc = lrelu(acc * (g[o] * rsqrtf(1.f + EPS_BN)) + bb[o]);
  out[t] = acc;
}

// ---------------------------------------------------------------------------
extern "C" void kernel_launch(void* const* d_in, const int* in_sizes, int n_in,
                              void* d_out, int out_size, void* d_ws, size_t ws_size,
                              hipStream_t stream) {
  (void)in_sizes; (void)n_in; (void)out_size; (void)ws_size;
  const float* x   = (const float*)d_in[0];
  const float* w1l = (const float*)d_in[1];  const float* g1l = (const float*)d_in[2];  const float* b1l = (const float*)d_in[3];
  const float* w1b = (const float*)d_in[4];  const float* g1b = (const float*)d_in[5];  const float* b1b = (const float*)d_in[6];
  const float* w1c = (const float*)d_in[7];  const float* g1c = (const float*)d_in[8];  const float* b1c = (const float*)d_in[9];
  const float* wf  = (const float*)d_in[10]; const float* gf  = (const float*)d_in[11]; const float* bf  = (const float*)d_in[12];
  const float* w2  = (const float*)d_in[13]; const float* g2  = (const float*)d_in[14]; const float* b2  = (const float*)d_in[15];
  const float* w3  = (const float*)d_in[16]; const float* g3  = (const float*)d_in[17]; const float* b3  = (const float*)d_in[18];
  const float* w4  = (const float*)d_in[19]; const float* g4  = (const float*)d_in[20]; const float* b4  = (const float*)d_in[21];
  const float* w5  = (const float*)d_in[22]; const float* g5  = (const float*)d_in[23]; const float* b5  = (const float*)d_in[24];
  const float* wl1 = (const float*)d_in[25]; const float* g6  = (const float*)d_in[26]; const float* b6  = (const float*)d_in[27];
  const float* wl2 = (const float*)d_in[28]; const float* bl2 = (const float*)d_in[29];
  const float* g7  = (const float*)d_in[30]; const float* b7  = (const float*)d_in[31];
  const float* wl3 = (const float*)d_in[32]; const float* bl3 = (const float*)d_in[33];

  const int M = NBATCH * NPTS;  // 32768 points

  // ---- workspace bump allocator (~220 MB; Gram buffer reused for x5) ----
  char* wsb = (char*)d_ws;
  size_t off = 0;
  auto take = [&](size_t bytes) -> char* {
    char* p = wsb + off;
    off = (off + bytes + 255) & ~((size_t)255);
    return p;
  };
  h16* wf16 = (h16*)take((size_t)64 * 192 * 2);
  h16* w516 = (h16*)take((size_t)1024 * 512 * 2);
  h16* wd2  = (h16*)take((size_t)64 * 64 * 2);   h16* wdf2 = (h16*)take((size_t)64 * 64 * 2);
  h16* wd3  = (h16*)take((size_t)128 * 64 * 2);  h16* wdf3 = (h16*)take((size_t)128 * 64 * 2);
  h16* wd4  = (h16*)take((size_t)256 * 128 * 2); h16* wdf4 = (h16*)take((size_t)256 * 128 * 2);
  int* idx  = (int*)take((size_t)M * 30 * 4);
  h16* xm   = (h16*)take((size_t)M * 192 * 2);
  h16* x1c  = (h16*)take((size_t)M * 64 * 2);
  h16* x2c  = (h16*)take((size_t)M * 64 * 2);
  h16* x3c  = (h16*)take((size_t)M * 128 * 2);
  h16* x4c  = (h16*)take((size_t)M * 256 * 2);
  h16* P    = (h16*)take((size_t)M * 256 * 2);
  h16* Q    = (h16*)take((size_t)M * 256 * 2);
  float* xx = (float*)take((size_t)M * 4);
  h16* G    = (h16*)take((size_t)NBATCH * NPTS * NPTS * 2);  // 134 MB
  h16* x5   = G;   // alias: G is dead after the last knn_feat, before conv5
  float* hb = (float*)take((size_t)NBATCH * 2048 * 4);
  float* h1 = (float*)take((size_t)NBATCH * 512 * 4);
  float* h2 = (float*)take((size_t)NBATCH * 256 * 4);

  auto blk = [](long long threads, int bs) { return dim3((unsigned)((threads + bs - 1) / bs)); };

  // ---- weight prep (f32 -> f16 / split Wd, Wc-Wd) ----
  cvt_f16<<<blk(64 * 192, 256), 256, 0, stream>>>(wf, wf16, 64 * 192);
  cvt_f16<<<blk(1024 * 512, 256), 256, 0, stream>>>(w5, w516, 1024 * 512);
  split_w<<<blk(64 * 64, 256), 256, 0, stream>>>(w2, wd2, wdf2, 64, 64);
  split_w<<<blk(128 * 64, 256), 256, 0, stream>>>(w3, wd3, wdf3, 128, 64);
  split_w<<<blk(256 * 128, 256), 256, 0, stream>>>(w4, wd4, wdf4, 256, 128);

  // ---- stage 0: xyz kNN (k=30) + hierarchical conv1 -> xm [M][192] ----
  knn_xyz<<<blk(M, 256), 256, 0, stream>>>(x, idx);
  hier_conv1<<<dim3(M), dim3(64), 0, stream>>>(x, idx, w1l, g1l, b1l,
                                               w1b, g1b, b1b, w1c, g1c, b1c, xm);

  // ---- convF: x1 = lrelu(bn(Wf * xm)), WMMA ----
  int nW = (M / 16) * (64 / 16);
  gemm_conv<<<blk((long long)nW * 32, 256), 256, 0, stream>>>(
      wf16, xm, x1c, gf, bf, 64, 192, 1, nW);

  const int nWg = NBATCH * (NPTS / 16) * (NPTS / 16);  // 262144 gram tiles

  // ---- edge stage 2 (Ci=64 -> Cout=64) ----
  gram_gemm<<<blk((long long)nWg * 32, 256), 256, 0, stream>>>(x1c, G, 64, nWg);
  rownorm<<<blk(M, 256), 256, 0, stream>>>(x1c, xx, 64, M);
  knn_feat<<<blk(M, 128), 128, 0, stream>>>(G, xx, idx);
  nW = (M / 16) * (64 / 16);
  gemm_conv<<<blk((long long)nW * 32, 256), 256, 0, stream>>>(wd2,  x1c, P, nullptr, nullptr, 64, 64, 0, nW);
  gemm_conv<<<blk((long long)nW * 32, 256), 256, 0, stream>>>(wdf2, x1c, Q, nullptr, nullptr, 64, 64, 0, nW);
  edge_combine<<<dim3(M), dim3(64), 0, stream>>>(P, Q, idx, g2, b2, x2c, 64);

  // ---- edge stage 3 (Ci=64 -> Cout=128) ----
  gram_gemm<<<blk((long long)nWg * 32, 256), 256, 0, stream>>>(x2c, G, 64, nWg);
  rownorm<<<blk(M, 256), 256, 0, stream>>>(x2c, xx, 64, M);
  knn_feat<<<blk(M, 128), 128, 0, stream>>>(G, xx, idx);
  nW = (M / 16) * (128 / 16);
  gemm_conv<<<blk((long long)nW * 32, 256), 256, 0, stream>>>(wd3,  x2c, P, nullptr, nullptr, 128, 64, 0, nW);
  gemm_conv<<<blk((long long)nW * 32, 256), 256, 0, stream>>>(wdf3, x2c, Q, nullptr, nullptr, 128, 64, 0, nW);
  edge_combine<<<dim3(M), dim3(128), 0, stream>>>(P, Q, idx, g3, b3, x3c, 128);

  // ---- edge stage 4 (Ci=128 -> Cout=256) ----
  gram_gemm<<<blk((long long)nWg * 32, 256), 256, 0, stream>>>(x3c, G, 128, nWg);
  rownorm<<<blk(M, 256), 256, 0, stream>>>(x3c, xx, 128, M);
  knn_feat<<<blk(M, 128), 128, 0, stream>>>(G, xx, idx);
  nW = (M / 16) * (256 / 16);
  gemm_conv<<<blk((long long)nW * 32, 256), 256, 0, stream>>>(wd4,  x3c, P, nullptr, nullptr, 256, 128, 0, nW);
  gemm_conv<<<blk((long long)nW * 32, 256), 256, 0, stream>>>(wdf4, x3c, Q, nullptr, nullptr, 256, 128, 0, nW);
  edge_combine<<<dim3(M), dim3(256), 0, stream>>>(P, Q, idx, g4, b4, x4c, 256);

  // ---- conv5 (1024 x 512, piecewise xcat) -> x5 [M][1024] (aliases G) ----
  int nW5 = (M / 16) * (1024 / 16);
  gemm_conv5<<<blk((long long)nW5 * 32, 256), 256, 0, stream>>>(
      w516, x1c, x2c, x3c, x4c, x5, g5, b5, nW5);

  // ---- pooling + FC head ----
  pool_kernel<<<blk(NBATCH * 1024, 256), 256, 0, stream>>>(x5, hb);
  fc_kernel<<<blk(NBATCH * 512, 256), 256, 0, stream>>>(hb, wl1, nullptr, g6, b6, h1, NBATCH, 2048, 512, 1);
  fc_kernel<<<blk(NBATCH * 256, 256), 256, 0, stream>>>(h1, wl2, bl2, g7, b7, h2, NBATCH, 512, 256, 1);
  fc_kernel<<<blk(NBATCH * 7, 112), 112, 0, stream>>>(h2, wl3, bl3, nullptr, nullptr, (float*)d_out, NBATCH, 256, 7, 0);
}
